// RotationEngine_63041529970724
// MI455X (gfx1250) — compile-verified, matrix operand order checked by reading, not run on hardware
//
#include <hip/hip_runtime.h>
#include <hip/hip_bf16.h>
#include <math.h>

#define B_     4
#define S_     2048
#define D_     1024
#define V_     32000
#define NCYC   28
#define NPL    64
#define RMS_EPS 1.1920929e-07f

typedef __attribute__((ext_vector_type(16))) __bf16 v16bf;
typedef __attribute__((ext_vector_type(8)))  float  v8f;

union FragBF { uint4 q[2]; v16bf v; };

__device__ __forceinline__ unsigned short f32_to_bf16(float f) {
    union { float f; unsigned int u; } x; x.f = f;
    unsigned int r = x.u + 0x7fffu + ((x.u >> 16) & 1u);   // round-to-nearest-even
    return (unsigned short)(r >> 16);
}

// gfx1250 async global->LDS copy, 16B per lane; tracked by ASYNCcnt.
// Flat pointers to LDS carry the LDS byte offset in their low 32 bits
// ({SHARED_BASE, offset} aperture encoding), so truncation gives VDST.
__device__ __forceinline__ void async_ld16(const unsigned short* lds_ptr,
                                           const unsigned short* g_ptr) {
    unsigned off = (unsigned)(size_t)lds_ptr;
    asm volatile("global_load_async_to_lds_b128 %0, %1, off"
                 :: "v"(off), "v"(g_ptr)
                 : "memory");
}

__device__ __forceinline__ void wait_async0() {
    asm volatile("s_wait_asynccnt 0x0" ::: "memory");
}

// ---------------------------------------------------------------------------
// 1) x[tok, :] = embed[tokens[tok], :]
// ---------------------------------------------------------------------------
__global__ void embed_gather(const int* __restrict__ tok,
                             const float* __restrict__ emb,
                             float* __restrict__ x) {
    int t   = blockIdx.x;                 // token index 0..B*S-1
    int id  = tok[t];
    const float4* src = (const float4*)(emb + (size_t)id * D_);
    float4*       dst = (float4*)(x + (size_t)t * D_);
    dst[threadIdx.x] = src[threadIdx.x];  // 256 threads * float4 = 1024
}

// ---------------------------------------------------------------------------
// 2) one "cycle": shift-mix, plane rotations, scale-shift-SiLU, RMSNorm(delta)
//    one block (256 threads) per token row of 1024 floats (float4/thread)
// ---------------------------------------------------------------------------
__global__ void cycle_kernel(const float* __restrict__ xin,
                             float* __restrict__ xout,
                             const float* __restrict__ ang,
                             const float* __restrict__ sc,
                             const float* __restrict__ sh,
                             const float* __restrict__ nw,
                             const float* __restrict__ mw,
                             const int*   __restrict__ pli,
                             const int*   __restrict__ plj,
                             int cyc) {
    __shared__ float row[D_];
    __shared__ float red[256];
    const int t   = threadIdx.x;
    const int tok = blockIdx.x;
    const int s   = tok & (S_ - 1);
    const size_t base = (size_t)tok * D_;

    float4 c = *(const float4*)(xin + base + t * 4);
    float4 l = make_float4(0.f, 0.f, 0.f, 0.f);
    float4 r = make_float4(0.f, 0.f, 0.f, 0.f);
    if (s > 0)      l = *(const float4*)(xin + base - D_ + t * 4);
    if (s < S_ - 1) r = *(const float4*)(xin + base + D_ + t * 4);

    const float m0 = mw[0], m1 = mw[1], m2 = mw[2];
    float4 xm;
    xm.x = m0 * l.x + m1 * c.x + m2 * r.x;
    xm.y = m0 * l.y + m1 * c.y + m2 * r.y;
    xm.z = m0 * l.z + m1 * c.z + m2 * r.z;
    xm.w = m0 * l.w + m1 * c.w + m2 * r.w;

    *(float4*)(row + t * 4) = xm;
    __syncthreads();

    // rotate 64 disjoint planes (read-all then write-all, barrier-separated)
    float ri = 0.f, rj = 0.f; int pi = 0, pj = 0;
    if (t < NPL) {
        pi = pli[t]; pj = plj[t];
        float a  = ang[(size_t)cyc * NPL + t];
        float ca = cosf(a), sa = sinf(a);
        float xi = row[pi], xj = row[pj];
        ri = xi * ca - xj * sa;
        rj = xi * sa + xj * ca;
    }
    __syncthreads();
    if (t < NPL) { row[pi] = ri; row[pj] = rj; }
    __syncthreads();

    float4 xr = *(const float4*)(row + t * 4);

    const float4 scv = *(const float4*)(sc + (size_t)cyc * D_ + t * 4);
    const float4 shv = *(const float4*)(sh + (size_t)cyc * D_ + t * 4);
    const float4 nwv = *(const float4*)(nw + (size_t)cyc * D_ + t * 4);

    float z0 = xr.x * scv.x + shv.x;
    float z1 = xr.y * scv.y + shv.y;
    float z2 = xr.z * scv.z + shv.z;
    float z3 = xr.w * scv.w + shv.w;
    float a0 = z0 / (1.f + expf(-z0));
    float a1 = z1 / (1.f + expf(-z1));
    float a2 = z2 / (1.f + expf(-z2));
    float a3 = z3 / (1.f + expf(-z3));
    float d0 = a0 - c.x, d1 = a1 - c.y, d2 = a2 - c.z, d3 = a3 - c.w;

    red[t] = d0 * d0 + d1 * d1 + d2 * d2 + d3 * d3;
    __syncthreads();
    for (int o = 128; o > 0; o >>= 1) {
        if (t < o) red[t] += red[t + o];
        __syncthreads();
    }
    float inv = rsqrtf(red[0] * (1.0f / D_) + RMS_EPS);

    float4 out;
    out.x = c.x + d0 * inv * nwv.x;
    out.y = c.y + d1 * inv * nwv.y;
    out.z = c.z + d2 * inv * nwv.z;
    out.w = c.w + d3 * inv * nwv.w;
    *(float4*)(xout + base + t * 4) = out;
}

// ---------------------------------------------------------------------------
// 3) final RMSNorm + bf16 pack of activations
// ---------------------------------------------------------------------------
__global__ void final_norm_bf16(const float* __restrict__ x,
                                const float* __restrict__ w,
                                unsigned short* __restrict__ xb) {
    __shared__ float red[256];
    const int t = threadIdx.x;
    const size_t base = (size_t)blockIdx.x * D_;
    float4 v = *(const float4*)(x + base + t * 4);
    red[t] = v.x * v.x + v.y * v.y + v.z * v.z + v.w * v.w;
    __syncthreads();
    for (int o = 128; o > 0; o >>= 1) {
        if (t < o) red[t] += red[t + o];
        __syncthreads();
    }
    float inv = rsqrtf(red[0] * (1.0f / D_) + RMS_EPS);
    float4 ww = *(const float4*)(w + t * 4);
    ushort4 o4;
    o4.x = f32_to_bf16(v.x * inv * ww.x);
    o4.y = f32_to_bf16(v.y * inv * ww.y);
    o4.z = f32_to_bf16(v.z * inv * ww.z);
    o4.w = f32_to_bf16(v.w * inv * ww.w);
    *(ushort4*)(xb + base + t * 4) = o4;
}

// ---------------------------------------------------------------------------
// 4) lm_head weights -> bf16 (one block per row)
// ---------------------------------------------------------------------------
__global__ void w_to_bf16(const float* __restrict__ w,
                          unsigned short* __restrict__ wb) {
    const size_t base = (size_t)blockIdx.x * D_ + threadIdx.x * 4;
    float4 v = *(const float4*)(w + base);
    ushort4 o4;
    o4.x = f32_to_bf16(v.x);
    o4.y = f32_to_bf16(v.y);
    o4.z = f32_to_bf16(v.z);
    o4.w = f32_to_bf16(v.w);
    *(ushort4*)(wb + base) = o4;
}

// ---------------------------------------------------------------------------
// 5) GEMM: C[M=8192, N=32000] = Xbf[M,K=1024] * Wbf[N,K]^T, f32 accumulate.
//    128x128 tile / block, 8 wave32 waves in 2x4; each wave: 4x2 16x16 frags.
//    BK=32, double-buffered LDS fed by GLOBAL_LOAD_ASYNC_TO_LDS_B128,
//    padded rows (40 bf16 = 80B stride, conflict-free b128 frag reads).
// ---------------------------------------------------------------------------
#define BM 128
#define BN 128
#define BK 32
#define LDP 40   // padded LDS row length in bf16 elements

__global__ __launch_bounds__(256) void gemm_bf16(
        const unsigned short* __restrict__ A,   // [8192 x 1024] bf16 row-major
        const unsigned short* __restrict__ Bw,  // [32000 x 1024] bf16 row-major
        float* __restrict__ C) {                // [8192 x 32000] f32
    constexpr int K  = D_;
    constexpr int NT = K / BK;                  // 32 k-steps

    __shared__ __align__(16) unsigned short As[2][BM * LDP];
    __shared__ __align__(16) unsigned short Bs[2][BM * LDP];

    const int t  = threadIdx.x;
    const int m0 = blockIdx.y * BM;
    const int n0 = blockIdx.x * BN;

    // global->LDS staging: thread t moves 16 bf16 (32B) of row lr, chunk lc
    const int lr = t >> 1;
    const int lc = (t & 1) << 4;                // 0 or 16
    const unsigned short* gA = A  + (size_t)(m0 + lr) * K + lc;
    const unsigned short* gB = Bw + (size_t)(n0 + lr) * K + lc;
    const int sOff = lr * LDP + lc;             // LDS element offset of chunk

    // wave mapping
    const int wave  = t >> 5;
    const int lane  = t & 31;
    const int wm    = wave & 1;                 // 0..1 -> 64-row M stripe
    const int wn    = wave >> 1;                // 0..3 -> 32-col N stripe
    const int row16 = lane & 15;                // M (or N) within 16x16 tile
    const int khalf = (lane >> 4) << 3;         // 0 or 8 (ISA A/B bf16 layout)

    v8f acc[4][2] = {};

    // async-stage k-step 0 into buffer 0
    async_ld16(&As[0][sOff],     gA);
    async_ld16(&As[0][sOff + 8], gA + 8);
    async_ld16(&Bs[0][sOff],     gB);
    async_ld16(&Bs[0][sOff + 8], gB + 8);
    wait_async0();
    __syncthreads();

#pragma unroll 1
    for (int kt = 0; kt < NT; ++kt) {
        const int cur = kt & 1;
        const int nxt = cur ^ 1;

        if (kt + 1 < NT) {
            const int k0 = (kt + 1) * BK;
            async_ld16(&As[nxt][sOff],     gA + k0);
            async_ld16(&As[nxt][sOff + 8], gA + k0 + 8);
            async_ld16(&Bs[nxt][sOff],     gB + k0);
            async_ld16(&Bs[nxt][sOff + 8], gB + k0 + 8);
        }
        if (kt + 2 < NT) {
            __builtin_prefetch(gA + (kt + 2) * BK, 0, 0);  // global_prefetch_b8
            __builtin_prefetch(gB + (kt + 2) * BK, 0, 0);
        }

        // build fragments per ISA 16-bit layout and issue 8 WMMAs
        FragBF fa[4], fb[2];
#pragma unroll
        for (int mi = 0; mi < 4; ++mi) {
            const int off = (wm * 64 + mi * 16 + row16) * LDP + khalf;
            fa[mi].q[0] = *(const uint4*)&As[cur][off];
            fa[mi].q[1] = *(const uint4*)&As[cur][off + 16];
        }
#pragma unroll
        for (int ni = 0; ni < 2; ++ni) {
            const int off = (wn * 32 + ni * 16 + row16) * LDP + khalf;
            fb[ni].q[0] = *(const uint4*)&Bs[cur][off];
            fb[ni].q[1] = *(const uint4*)&Bs[cur][off + 16];
        }
#pragma unroll
        for (int mi = 0; mi < 4; ++mi)
#pragma unroll
            for (int ni = 0; ni < 2; ++ni)
                acc[mi][ni] = __builtin_amdgcn_wmma_f32_16x16x32_bf16(
                    false, fa[mi].v, false, fb[ni].v,
                    (short)0, acc[mi][ni], false, false);

        if (kt + 1 < NT) wait_async0();   // next buffer fully in LDS
        __syncthreads();
    }

    // epilogue: C/D layout -> VGPR r holds M = r + (lane>>4)*8, N = lane&15
#pragma unroll
    for (int mi = 0; mi < 4; ++mi) {
#pragma unroll
        for (int ni = 0; ni < 2; ++ni) {
            const int n = n0 + wn * 32 + ni * 16 + (lane & 15);
            const int mb = m0 + wm * 64 + mi * 16 + ((lane >> 4) << 3);
#pragma unroll
            for (int rr = 0; rr < 8; ++rr)
                C[(size_t)(mb + rr) * V_ + n] = acc[mi][ni][rr];
        }
    }
}

// ---------------------------------------------------------------------------
// launcher
// ---------------------------------------------------------------------------
extern "C" void kernel_launch(void* const* d_in, const int* in_sizes, int n_in,
                              void* d_out, int out_size, void* d_ws, size_t ws_size,
                              hipStream_t stream) {
    (void)in_sizes; (void)n_in; (void)out_size; (void)ws_size;
    const int*   tokens = (const int*)  d_in[0];
    const float* embed  = (const float*)d_in[1];
    const float* lmw    = (const float*)d_in[2];
    const float* onw    = (const float*)d_in[3];
    const float* ang    = (const float*)d_in[4];
    const float* sc     = (const float*)d_in[5];
    const float* sh     = (const float*)d_in[6];
    const float* mw     = (const float*)d_in[7];
    const float* nw     = (const float*)d_in[8];
    const int*   pli    = (const int*)  d_in[9];
    const int*   plj    = (const int*)  d_in[10];
    float* out = (float*)d_out;

    char* ws = (char*)d_ws;
    const size_t XBYTES = (size_t)B_ * S_ * D_ * sizeof(float);       // 32 MB
    float*          x0 = (float*)ws;
    float*          x1 = (float*)(ws + XBYTES);
    unsigned short* xb = (unsigned short*)(ws + 2 * XBYTES);          // 16 MB
    unsigned short* wb = (unsigned short*)(ws + 2 * XBYTES + XBYTES / 2);

    const int NTOK = B_ * S_;

    embed_gather<<<NTOK, 256, 0, stream>>>(tokens, embed, x0);

    for (int c = 0; c < NCYC; ++c) {
        const float* in   = (c & 1) ? x1 : x0;
        float*       outp = (c & 1) ? x0 : x1;
        cycle_kernel<<<NTOK, 256, 0, stream>>>(in, outp, ang, sc, sh, nw,
                                               mw, pli, plj, c);
    }
    // after 28 cycles (even count) result is back in x0

    final_norm_bf16<<<NTOK, 256, 0, stream>>>(x0, onw, xb);
    w_to_bf16<<<V_, 256, 0, stream>>>(lmw, wb);

    dim3 grid(V_ / BN, (B_ * S_) / BM);   // 250 x 64
    gemm_bf16<<<grid, 256, 0, stream>>>(xb, wb, out);
}
